// CausalSelfAttention_62096637165886
// MI455X (gfx1250) — matrix operand order repaired and checked by reading, not compiled, not run on hardware
//
#include <hip/hip_runtime.h>

// ---------------------------------------------------------------------------
// Causal self-attention forward for MI455X (gfx1250, wave32, WMMA).
// All matmuls run on v_wmma_f32_16x16x32_bf16 (fp32 accum).
// Problem: S=2048, B=2, D=768, H=12, hd=64.  ~45 GFLOP dense (~32 causal), ~35 MB of tensors
// (fits in 192MB L2) -> matrix-core bound, so bf16 WMMA everywhere.
// GEMMs stage the shared B tile through LDS with global_load_async_to_lds
// (ASYNCcnt) in a double-buffered pipeline so WMMAs overlap the copies.
// ---------------------------------------------------------------------------

typedef __attribute__((ext_vector_type(16))) __bf16       v16bf;
typedef __attribute__((ext_vector_type(8)))  float        v8f;
typedef __attribute__((ext_vector_type(4)))  unsigned int v4u;

#define S_SEQ  2048
#define BATCH  2
#define DMODEL 768
#define NHEADS 12
#define HDIM   64
#define MROWS  (S_SEQ * BATCH)   // 4096 tokens (row = s*B + b)

union FragU { v16bf f; v4u q[2]; };

// A-matrix 16x32 bf16 fragment (ISA 7.12.2): lane r=lane%16 holds row r;
// lanes 0-15: K=0..7 in v0-3, K=16..23 in v4-7; lanes 16-31: +8 on each.
__device__ __forceinline__ v16bf load_frag_a(const __bf16* base, int stride) {
  const int lane = threadIdx.x & 31;
  const int r    = lane & 15;
  const int half = lane >> 4;
  const __bf16* p = base + (size_t)r * stride;
  FragU u;
  u.q[0] = *(const v4u*)(p + 8 * half);
  u.q[1] = *(const v4u*)(p + 16 + 8 * half);
  return u.f;
}

// B-matrix 32x16 bf16 fragment: lane n=lane%16 holds column n;
// lanes 0-15: K=0..7 in v0-3, K=16..23 in v4-7; lanes 16-31: +8 on each.
// `base` points at Bt (the N x K transposed view), row-major, pitch=stride.
__device__ __forceinline__ v16bf load_frag_b(const __bf16* base, int stride) {
  const int lane = threadIdx.x & 31;
  const int n    = lane & 15;
  const int half = lane >> 4;
  const __bf16* p = base + (size_t)n * stride;
  FragU u;
  u.q[0] = *(const v4u*)(p + 8 * half);
  u.q[1] = *(const v4u*)(p + 16 + 8 * half);
  return u.f;
}

__device__ __forceinline__ v8f wmma_bf16(v16bf a, v16bf b, v8f c) {
  return __builtin_amdgcn_wmma_f32_16x16x32_bf16(
      /*neg_a=*/false, a, /*neg_b=*/false, b,
      /*c_mod=*/(short)0, c, /*reuse_a=*/false, /*reuse_b=*/false);
}

// Async copy 16B global -> LDS (per lane).  Tracked with ASYNCcnt.
__device__ __forceinline__ void async_b128(unsigned lds_off, const void* g) {
  asm volatile("global_load_async_to_lds_b128 %0, %1, off"
               :: "v"(lds_off), "v"((unsigned long long)g) : "memory");
}
__device__ __forceinline__ void wait_async(int n) {
  if (n == 0) asm volatile("s_wait_asynccnt 0x0" ::: "memory");
  else        asm volatile("s_wait_asynccnt 0x4" ::: "memory");
}
// LDS aperture address: low 32 bits of the generic pointer (ISA 10.2).
__device__ __forceinline__ unsigned lds_addr(const void* p) {
  return (unsigned)(unsigned long long)p;
}

// ---------------------------------------------------------------------------
// Conversion / transpose helpers
// ---------------------------------------------------------------------------
__global__ void cvt_f32_bf16(const float* __restrict__ src,
                             __bf16* __restrict__ dst, int nElem) {
  int i = blockIdx.x * blockDim.x + threadIdx.x;
  if (i < nElem) dst[i] = (__bf16)src[i];
}

// W (R x C) fp32  ->  Wt (C x R) bf16
__global__ void transpose_cvt(const float* __restrict__ W,
                              __bf16* __restrict__ Wt, int R, int C) {
  int i = blockIdx.x * blockDim.x + threadIdx.x;
  if (i < R * C) {
    int r = i / C, c = i % C;
    Wt[(size_t)c * R + r] = (__bf16)W[(size_t)r * C + c];
  }
}

// V (token-major, [MROWS][DMODEL]) bf16 -> Vt [B][H][HDIM][S] bf16
__global__ void transpose_v(const __bf16* __restrict__ V,
                            __bf16* __restrict__ Vt) {
  int i = blockIdx.x * blockDim.x + threadIdx.x;
  const int total = BATCH * NHEADS * HDIM * S_SEQ;
  if (i >= total) return;
  int t = i % S_SEQ; int j = i / S_SEQ;
  int d = j % HDIM;  j /= HDIM;
  int h = j % NHEADS; int b = j / NHEADS;
  Vt[i] = V[((size_t)t * BATCH + b) * DMODEL + h * HDIM + d];
}

// ---------------------------------------------------------------------------
// WMMA GEMM:  C(MxN) = A(MxK,bf16) * Bt(NxK,bf16)^T + bias
// Block = 128 threads = 4 waves; wave computes a 16x64 strip (4 accumulators).
// The 64x64(K-slice) B tile is shared by all waves: staged into LDS with
// async-to-LDS copies, double buffered over K chunks of 64.
// ---------------------------------------------------------------------------
__global__ void __launch_bounds__(128) gemm_bf16_wmma(
    const __bf16* __restrict__ A, const __bf16* __restrict__ Bt,
    const float* __restrict__ bias, float* __restrict__ Cf,
    __bf16* __restrict__ Cb, int M, int N, int K) {
  __shared__ __align__(16) __bf16 bufs[2][64][64];  // 2 x 8KB

  const int wave = threadIdx.x >> 5;
  const int lane = threadIdx.x & 31;
  const int n    = lane & 15;
  const int half = lane >> 4;
  const int m0 = blockIdx.x * 64 + wave * 16;
  const int n0 = blockIdx.y * 64;
  const int tid = threadIdx.x;

  // stage one 64x64 bf16 chunk of Bt (rows n0..n0+63, cols k0..k0+63)
  auto stage = [&](int k0, int bufIdx) {
    const unsigned base = lds_addr(&bufs[bufIdx][0][0]);
#pragma unroll
    for (int t = 0; t < 4; ++t) {               // 4 async b128 per lane
      const int c    = t * 128 + tid;           // 0..511
      const int row  = c >> 3;
      const int col8 = (c & 7) * 8;
      async_b128(base + (unsigned)(row * 64 + col8) * 2u,
                 Bt + (size_t)(n0 + row) * K + k0 + col8);
    }
  };

  v8f acc[4] = {{}, {}, {}, {}};
  const int nch = K >> 6;                       // K/64 chunks
  stage(0, 0);
  for (int c = 0; c < nch; ++c) {
    if (c + 1 < nch) { stage((c + 1) << 6, (c + 1) & 1); wait_async(4); }
    else             { wait_async(0); }
    __syncthreads();                            // chunk c visible to all waves
    const __bf16* bb = &bufs[c & 1][0][0];
#pragma unroll
    for (int ks = 0; ks < 2; ++ks) {            // two K-steps of 32
      v16bf a = load_frag_a(A + (size_t)m0 * K + (c << 6) + ks * 32, K);
#pragma unroll
      for (int t = 0; t < 4; ++t) {
        v16bf b = load_frag_b(bb + (16 * t) * 64 + ks * 32, 64);
        acc[t] = wmma_bf16(a, b, acc[t]);
      }
    }
    __syncthreads();                            // done reading before reuse
  }

#pragma unroll
  for (int t = 0; t < 4; ++t) {
#pragma unroll
    for (int j = 0; j < 8; ++j) {
      const int row = m0 + j + 8 * half;        // C/D layout: VGPR j, lane half
      const int col = n0 + 16 * t + n;
      float v = acc[t][j] + bias[col];
      if (Cf) Cf[(size_t)row * N + col] = v;
      else    Cb[(size_t)row * N + col] = (__bf16)v;
    }
  }
}

// ---------------------------------------------------------------------------
// Flash-attention forward (causal).  Block = 128 thr = 4 waves; each wave owns
// a 16-query tile of one (b,h).  Key blocks of 32: 4 WMMA for QK^T, online
// softmax (f32, shfl_xor butterfly over the 16 lanes holding one row), P
// staged via LDS to re-layout C->A fragment, 4 WMMA for P*V against Vt.
// ---------------------------------------------------------------------------
__global__ void __launch_bounds__(128) attn_fwd_wmma(
    const __bf16* __restrict__ Q, const __bf16* __restrict__ Kmat,
    const __bf16* __restrict__ Vt, __bf16* __restrict__ O) {
  __shared__ __align__(16) __bf16 plds[4][16][32];
  const int wave = threadIdx.x >> 5;
  const int lane = threadIdx.x & 31;
  const int n    = lane & 15;
  const int half = lane >> 4;
  const int h = blockIdx.y % NHEADS;
  const int b = blockIdx.y / NHEADS;
  const int qbase = blockIdx.x * 64 + wave * 16;

  const int rowstride = BATCH * DMODEL;  // 1536: consecutive s for fixed b
  const __bf16* qp = Q + ((size_t)qbase * BATCH + b) * DMODEL + h * HDIM;
  const v16bf qa0 = load_frag_a(qp, rowstride);       // head dims  0..31
  const v16bf qa1 = load_frag_a(qp + 32, rowstride);  // head dims 32..63

  v8f acc[4] = {{}, {}, {}, {}};
  float mrow[8], lrow[8];
#pragma unroll
  for (int j = 0; j < 8; ++j) { mrow[j] = -1e30f; lrow[j] = 0.f; }

  const int nkb = (qbase + 16 + 31) >> 5;  // key blocks of 32 covering <= qbase+15
  for (int ib = 0; ib < nkb; ++ib) {
    const int kk = ib * 32;

    // ---- scores: two 16x16 sub-tiles (keys kk..+15, kk+16..+31) ----
    v8f sc[2];
#pragma unroll
    for (int c = 0; c < 2; ++c) {
      const __bf16* kp =
          Kmat + ((size_t)(kk + 16 * c) * BATCH + b) * DMODEL + h * HDIM;
      v16bf b0 = load_frag_b(kp, rowstride);       // dims  0..31 of K rows
      v16bf b1 = load_frag_b(kp + 32, rowstride);  // dims 32..63
      v8f s = {};
      s = wmma_bf16(qa0, b0, s);
      s = wmma_bf16(qa1, b1, s);
      sc[c] = s;
    }

    // ---- causal mask + online softmax ----
    float alpha[8];
#pragma unroll
    for (int j = 0; j < 8; ++j) {
      const int srow = qbase + j + 8 * half;      // query index of this row
      float v0 = sc[0][j] * 0.125f;               // 1/sqrt(64)
      float v1 = sc[1][j] * 0.125f;
      if (kk + n > srow)      v0 = -1e30f;
      if (kk + 16 + n > srow) v1 = -1e30f;
      float rm = fmaxf(v0, v1);
#pragma unroll
      for (int msk = 1; msk < 16; msk <<= 1)
        rm = fmaxf(rm, __shfl_xor(rm, msk, 32));
      const float mn = fmaxf(mrow[j], rm);
      alpha[j] = __expf(mrow[j] - mn);
      mrow[j] = mn;
      const float e0 = __expf(v0 - mn);
      const float e1 = __expf(v1 - mn);
      float rs = e0 + e1;
#pragma unroll
      for (int msk = 1; msk < 16; msk <<= 1)
        rs += __shfl_xor(rs, msk, 32);
      lrow[j] = lrow[j] * alpha[j] + rs;
      // stage P (C layout -> row-major LDS tile) for the A-fragment reload
      plds[wave][j + 8 * half][n]      = (__bf16)e0;
      plds[wave][j + 8 * half][16 + n] = (__bf16)e1;
    }
#pragma unroll
    for (int t = 0; t < 4; ++t)
#pragma unroll
      for (int j = 0; j < 8; ++j) acc[t][j] *= alpha[j];

    asm volatile("s_wait_dscnt 0x0" ::: "memory");  // RAW through LDS (per-wave)
    const v16bf pa = load_frag_a(&plds[wave][0][0], 32);

    // ---- P(16x32) @ V(32x64): 4 n-tiles against pre-transposed Vt ----
#pragma unroll
    for (int t = 0; t < 4; ++t) {
      const __bf16* vp =
          Vt + (((size_t)b * NHEADS + h) * HDIM + t * 16) * S_SEQ + kk;
      v16bf vb = load_frag_b(vp, S_SEQ);
      acc[t] = wmma_bf16(pa, vb, acc[t]);
    }
  }

  // ---- normalize + store attention output (bf16, token-major) ----
#pragma unroll
  for (int t = 0; t < 4; ++t)
#pragma unroll
    for (int j = 0; j < 8; ++j) {
      const int s = qbase + j + 8 * half;
      const float o = acc[t][j] / lrow[j];
      O[((size_t)s * BATCH + b) * DMODEL + h * HDIM + t * 16 + n] = (__bf16)o;
    }
}

// ---------------------------------------------------------------------------
// Host launcher
// ---------------------------------------------------------------------------
extern "C" void kernel_launch(void* const* d_in, const int* in_sizes, int n_in,
                              void* d_out, int out_size, void* d_ws,
                              size_t ws_size, hipStream_t stream) {
  const float* x  = (const float*)d_in[0];
  const float* Wq = (const float*)d_in[1];
  const float* bq = (const float*)d_in[2];
  const float* Wk = (const float*)d_in[3];
  const float* bk = (const float*)d_in[4];
  const float* Wv = (const float*)d_in[5];
  const float* bv = (const float*)d_in[6];
  const float* Wo = (const float*)d_in[7];
  const float* bo = (const float*)d_in[8];
  float* out = (float*)d_out;

  const size_t XN = (size_t)MROWS * DMODEL;   // 3,145,728 elements
  const size_t WN = (size_t)DMODEL * DMODEL;  //   589,824 elements

  char* ws = (char*)d_ws;
  __bf16* xb  = (__bf16*)ws; ws += XN * 2;
  __bf16* WqT = (__bf16*)ws; ws += WN * 2;
  __bf16* WkT = (__bf16*)ws; ws += WN * 2;
  __bf16* WvT = (__bf16*)ws; ws += WN * 2;
  __bf16* WoT = (__bf16*)ws; ws += WN * 2;
  __bf16* Qb  = (__bf16*)ws; ws += XN * 2;
  __bf16* Kb  = (__bf16*)ws; ws += XN * 2;
  __bf16* Vb  = (__bf16*)ws; ws += XN * 2;
  __bf16* Vtr = (__bf16*)ws; ws += XN * 2;
  __bf16* Yb  = (__bf16*)ws; ws += XN * 2;

  const int thr = 256;
  cvt_f32_bf16<<<(int)((XN + thr - 1) / thr), thr, 0, stream>>>(x, xb, (int)XN);
  transpose_cvt<<<(int)((WN + thr - 1) / thr), thr, 0, stream>>>(Wq, WqT, DMODEL, DMODEL);
  transpose_cvt<<<(int)((WN + thr - 1) / thr), thr, 0, stream>>>(Wk, WkT, DMODEL, DMODEL);
  transpose_cvt<<<(int)((WN + thr - 1) / thr), thr, 0, stream>>>(Wv, WvT, DMODEL, DMODEL);
  transpose_cvt<<<(int)((WN + thr - 1) / thr), thr, 0, stream>>>(Wo, WoT, DMODEL, DMODEL);

  dim3 ggrid(MROWS / 64, DMODEL / 64);  // 64 x 12 blocks
  gemm_bf16_wmma<<<ggrid, 128, 0, stream>>>(xb, WqT, bq, nullptr, Qb,
                                            MROWS, DMODEL, DMODEL);
  gemm_bf16_wmma<<<ggrid, 128, 0, stream>>>(xb, WkT, bk, nullptr, Kb,
                                            MROWS, DMODEL, DMODEL);
  gemm_bf16_wmma<<<ggrid, 128, 0, stream>>>(xb, WvT, bv, nullptr, Vb,
                                            MROWS, DMODEL, DMODEL);

  transpose_v<<<(int)((XN + thr - 1) / thr), thr, 0, stream>>>(Vb, Vtr);

  dim3 agrid(S_SEQ / 64, BATCH * NHEADS);  // 32 x 24 blocks, 4 waves each
  attn_fwd_wmma<<<agrid, 128, 0, stream>>>(Qb, Kb, Vtr, Yb);

  gemm_bf16_wmma<<<ggrid, 128, 0, stream>>>(Yb, WoT, bo, out, nullptr,
                                            MROWS, DMODEL, DMODEL);
}
